// HiLo_SelfAttention_21371757265021
// MI455X (gfx1250) — matrix operand order repaired and checked by reading, not compiled, not run on hardware
//
#include <hip/hip_runtime.h>
#include <hip/hip_bf16.h>
#include <math.h>

typedef __attribute__((ext_vector_type(16))) _Float16 v16h;
typedef __attribute__((ext_vector_type(8)))  float    v8f;

// ---------------------------------------------------------------------------
// WMMA helpers (CDNA5 / gfx1250, wave32)
// ---------------------------------------------------------------------------
__device__ inline v8f wmma32(v16h a, v16h b, v8f c) {
  // D = A(16x32 f16) x B(32x16 f16) + C(16x16 f32)
  return __builtin_amdgcn_wmma_f32_16x16x32_f16(
      /*neg_a=*/false, a, /*neg_b=*/false, b,
      /*c_mod=*/(short)0, c, /*reuse_a=*/false, /*reuse_b=*/false);
}

// Load A fragment (M x K tile, 16x32) from a row-major [rows][ld] f16 buffer.
// Layout per ISA 7.12.2: lane m=lane&15; VGPR v: k=(v>>2)*16 + (lane>>4)*8 + (v&3)*2.
__device__ inline v16h load_frag_mk(const _Float16* p, int ld, int r0, int k0) {
  int lane = threadIdx.x & 31;
  const unsigned short* row = (const unsigned short*)(p + (long)(r0 + (lane & 15)) * ld);
  int kh = (lane >> 4) * 8;
  union { v16h v; unsigned u[8]; } f;
#pragma unroll
  for (int i = 0; i < 8; ++i) {
    int k = k0 + ((i >> 2) << 4) + kh + ((i & 3) << 1);
    f.u[i] = *(const unsigned*)(row + k);
  }
  return f.v;
}

// Load B fragment (K x N tile, 32x16) from B stored transposed: row-major [N][ld(K)].
// Layout: lane col n=lane&15; k = (v>>2)*16 + (lane>>4)*8 + (v&3)*2.
__device__ inline v16h load_frag_nk(const _Float16* p, int ld, int n0, int k0) {
  int lane = threadIdx.x & 31;
  const unsigned short* row = (const unsigned short*)(p + (long)(n0 + (lane & 15)) * ld);
  int kb = k0 + ((lane >> 4) << 3);
  union { v16h v; unsigned u[8]; } f;
#pragma unroll
  for (int i = 0; i < 8; ++i) f.u[i] = *(const unsigned*)(row + kb + ((i >> 2) << 4) + ((i & 3) << 1));
  return f.v;
}

// ---------------------------------------------------------------------------
// Prep: convert the six 256x256 weights to f16; fold ConvTranspose + 1x1 conv
// into W2[ky*2+kx] (64x64 f16) ; W1 = po_w[:, :64] ; bias = po_w[:,64:] @ ct_b
// ---------------------------------------------------------------------------
__global__ void prep_kernel(const float* __restrict__ wq_hi, const float* __restrict__ wk_hi,
                            const float* __restrict__ wv_hi, const float* __restrict__ wq_lo,
                            const float* __restrict__ wk_lo, const float* __restrict__ wv_lo,
                            const float* __restrict__ ct_w, const float* __restrict__ ct_b,
                            const float* __restrict__ po_w,
                            _Float16* __restrict__ w16, _Float16* __restrict__ w1,
                            _Float16* __restrict__ w2, float* __restrict__ bias) {
  const int NCV = 6 * 65536;
  long tid = (long)blockIdx.x * blockDim.x + threadIdx.x;
  if (tid < NCV) {
    int mi = (int)(tid >> 16), e = (int)(tid & 65535);
    const float* src = (mi == 0) ? wq_hi : (mi == 1) ? wk_hi : (mi == 2) ? wv_hi
                     : (mi == 3) ? wq_lo : (mi == 4) ? wk_lo : wv_lo;
    w16[tid] = (_Float16)src[e];
  } else if (tid < NCV + 4096) {
    int e = (int)(tid - NCV);
    int o = e >> 6, i = e & 63;
    w1[e] = (_Float16)po_w[o * 128 + i];
  } else if (tid < NCV + 4096 + 16384) {
    int e = (int)(tid - NCV - 4096);
    int par = e >> 12;            // ky*2+kx
    int o = (e >> 6) & 63, ci = e & 63;
    int ky = par >> 1, kx = par & 1;
    float s = 0.f;
    for (int c = 0; c < 64; ++c)
      s += po_w[o * 128 + 64 + c] * ct_w[((ci * 64 + c) * 2 + ky) * 2 + kx];
    w2[e] = (_Float16)s;          // layout [par][o][ci]
  } else if (tid < NCV + 4096 + 16384 + 64) {
    int o = (int)(tid - NCV - 4096 - 16384);
    float s = 0.f;
    for (int c = 0; c < 64; ++c) s += po_w[o * 128 + 64 + c] * ct_b[c];
    bias[o] = s;
  }
}

// ---------------------------------------------------------------------------
// 2x2 max pool: [2,64,384,384] -> [2,64,192,192]
// ---------------------------------------------------------------------------
__global__ void maxpool_kernel(const float* __restrict__ x, float* __restrict__ x_lo) {
  int t = blockIdx.x * 256 + threadIdx.x;           // 2*64*192*192 = 4718592 exact
  int j = t % 192;
  int i = (t / 192) % 192;
  long bc = t / (192 * 192);
  const float* p = x + (bc * 384 + 2 * i) * 384 + 2 * j;
  x_lo[t] = fmaxf(fmaxf(p[0], p[1]), fmaxf(p[384], p[385]));
}

// ---------------------------------------------------------------------------
// Per-token self-attention. 1 workgroup = 1 token; 512 threads = 16 waves.
// LDS: attn[256][256] f16 (first 32KB aliased as X[c][p]) + Qn[p][c] + Kn[q][c]
//      + Vt[c][q]  => 224 KB
// ---------------------------------------------------------------------------
__global__ __launch_bounds__(512) void attn_kernel(
    const float* __restrict__ x, const _Float16* __restrict__ wq,
    const _Float16* __restrict__ wk, const _Float16* __restrict__ wv,
    const float* __restrict__ temp_p, float* __restrict__ img,
    int H, int W, int hb, int wb) {
  __shared__ _Float16 sA[256 * 256];   // attn [p][q]; first 64*256 doubles as X[c][p]
  __shared__ _Float16 sQ[256 * 64];    // qn [p][c]
  __shared__ _Float16 sK[256 * 64];    // kn [q][c]
  __shared__ _Float16 sV[64 * 256];    // v  [c][q]  (transposed for B fragments)
  _Float16* sX = sA;

  const int tid = threadIdx.x;
  const int lane = tid & 31;
  const int wid = tid >> 5;
  const int ntok = hb * wb;
  const int n = blockIdx.x % ntok;
  const int bb = blockIdx.x / ntok;
  const int ti = n / wb, tj = n % wb;
  const float temp = temp_p[0];

  // ---- gather X tile: element (c, p=ph*16+pw) at x[b, c, ph*hb+ti, pw*wb+tj]
  for (int t = tid; t < 64 * 256; t += 512) {
    int c = t >> 8, p = t & 255;
    int ph = p >> 4, pw = p & 15;
    sX[t] = (_Float16)x[(((long)bb * 64 + c) * H + ph * hb + ti) * W + pw * wb + tj];
  }
  __syncthreads();

  // ---- Q,K,V = W @ X^T :  M=q(256, 16 tiles), N=c(64, 4 tiles), K=p(256)
  {
    v8f aq[4] = {}, ak[4] = {}, av[4] = {};
    const int m0 = wid * 16;
    for (int kk = 0; kk < 8; ++kk) {
      const int k0 = kk * 32;
      v16h b[4];
#pragma unroll
      for (int nt = 0; nt < 4; ++nt) b[nt] = load_frag_nk(sX, 256, nt * 16, k0);
      v16h a = load_frag_mk(wq, 256, m0, k0);
#pragma unroll
      for (int nt = 0; nt < 4; ++nt) aq[nt] = wmma32(a, b[nt], aq[nt]);
      a = load_frag_mk(wk, 256, m0, k0);
#pragma unroll
      for (int nt = 0; nt < 4; ++nt) ak[nt] = wmma32(a, b[nt], ak[nt]);
      a = load_frag_mk(wv, 256, m0, k0);
#pragma unroll
      for (int nt = 0; nt < 4; ++nt) av[nt] = wmma32(a, b[nt], av[nt]);
    }
    const int colc = lane & 15, kh = lane >> 4;
#pragma unroll
    for (int nt = 0; nt < 4; ++nt) {
      int c = nt * 16 + colc;
#pragma unroll
      for (int r = 0; r < 8; ++r) {
        int q = m0 + r + 8 * kh;
        sQ[q * 64 + c] = (_Float16)aq[nt][r];
        sK[q * 64 + c] = (_Float16)ak[nt][r];
        sV[c * 256 + q] = (_Float16)av[nt][r];   // store V transposed
      }
    }
  }
  __syncthreads();

  // ---- L2 normalize rows of Q and K (512 threads -> 512 rows)
  {
    _Float16* base = (tid < 256) ? (sQ + tid * 64) : (sK + (tid - 256) * 64);
    float ss = 0.f;
    for (int c = 0; c < 64; ++c) { float v = (float)base[c]; ss += v * v; }
    float sc = 1.0f / fmaxf(sqrtf(ss), 1e-12f);
    for (int c = 0; c < 64; ++c) base[c] = (_Float16)((float)base[c] * sc);
  }
  __syncthreads();

  // ---- S = Qn @ Kn^T, softmax over q; wave owns a full 16-row stripe
  {
    v8f s[16] = {};
    const int m0 = wid * 16;
    for (int kk = 0; kk < 2; ++kk) {
      v16h a = load_frag_mk(sQ, 64, m0, kk * 32);
#pragma unroll
      for (int nt = 0; nt < 16; ++nt) {
        v16h b = load_frag_nk(sK, 64, nt * 16, kk * 32);
        s[nt] = wmma32(a, b, s[nt]);
      }
    }
    const int colq = lane & 15, kh = lane >> 4;
#pragma unroll
    for (int r = 0; r < 8; ++r) {
      float mx = -3.4e38f;
#pragma unroll
      for (int nt = 0; nt < 16; ++nt) {
        float z = s[nt][r] * temp; s[nt][r] = z; mx = fmaxf(mx, z);
      }
      for (int off = 1; off < 16; off <<= 1) mx = fmaxf(mx, __shfl_xor(mx, off, 32));
      float sum = 0.f;
#pragma unroll
      for (int nt = 0; nt < 16; ++nt) {
        float e = __expf(s[nt][r] - mx); s[nt][r] = e; sum += e;
      }
      for (int off = 1; off < 16; off <<= 1) sum += __shfl_xor(sum, off, 32);
      float inv = 1.0f / sum;
      int m = m0 + r + 8 * kh;
#pragma unroll
      for (int nt = 0; nt < 16; ++nt)
        sA[m * 256 + nt * 16 + colq] = (_Float16)(s[nt][r] * inv);
    }
  }
  __syncthreads();

  // ---- O = attn @ V ; scatter-unpatchify to img[b,c,ph*hb+ti,pw*wb+tj]
  {
    v8f o[4] = {};
    const int m0 = wid * 16;
    for (int kk = 0; kk < 8; ++kk) {
      v16h a = load_frag_mk(sA, 256, m0, kk * 32);
#pragma unroll
      for (int nt = 0; nt < 4; ++nt) {
        v16h b = load_frag_nk(sV, 256, nt * 16, kk * 32);
        o[nt] = wmma32(a, b, o[nt]);
      }
    }
    const int colc = lane & 15, kh = lane >> 4;
#pragma unroll
    for (int nt = 0; nt < 4; ++nt) {
      int c = nt * 16 + colc;
#pragma unroll
      for (int r = 0; r < 8; ++r) {
        int p = m0 + r + 8 * kh;
        int ph = p >> 4, pw = p & 15;
        img[(((long)bb * 64 + c) * H + ph * hb + ti) * W + pw * wb + tj] = o[nt][r];
      }
    }
  }
}

// ---------------------------------------------------------------------------
// Epilogue: out = W1 @ img_hi + W2[h&1][w&1] @ img_lo(up) + bias, all WMMA.
// 1 workgroup = (b, row h, 64-wide w block); pixels reordered by w-parity so
// each 16-pixel N-tile uses a single ConvTranspose tap.
// ---------------------------------------------------------------------------
__global__ __launch_bounds__(256) void epilogue_kernel(
    const float* __restrict__ img_hi, const float* __restrict__ img_lo,
    const _Float16* __restrict__ w1, const _Float16* __restrict__ w2,
    const float* __restrict__ bias, float* __restrict__ out) {
  __shared__ _Float16 sHi[64 * 64];   // [pixslot][c], slots: 0-31 even w, 32-63 odd w
  __shared__ _Float16 sLo[32 * 64];   // [lo-pix][c]

  const int tid = threadIdx.x, lane = tid & 31, wid = tid >> 5;
  const int bid = blockIdx.x;
  const int wblk = bid % 6;
  const int h = (bid / 6) % 384;
  const int b = bid / (6 * 384);
  const int w0 = wblk * 64;

  for (int t = tid; t < 64 * 64; t += 256) {
    int c = t >> 6, ww = t & 63;
    float v = img_hi[(((long)b * 64 + c) * 384 + h) * 384 + w0 + ww];
    int s = (ww & 1) * 32 + (ww >> 1);
    sHi[s * 64 + c] = (_Float16)v;
  }
  const int hl = h >> 1, wl0 = w0 >> 1;
  for (int t = tid; t < 32 * 64; t += 256) {
    int c = t >> 5, wwl = t & 31;
    float v = img_lo[(((long)b * 64 + c) * 192 + hl) * 192 + wl0 + wwl];
    sLo[wwl * 64 + c] = (_Float16)v;
  }
  __syncthreads();

  const int hy = h & 1;
#pragma unroll
  for (int tno = 0; tno < 2; ++tno) {
    int tileid = wid * 2 + tno;          // 16 tiles over 8 waves
    int mt = tileid >> 2, nt = tileid & 3;
    v8f acc = {};
    for (int kk = 0; kk < 2; ++kk) {     // hi chain, K = 64
      v16h a = load_frag_mk(w1, 64, mt * 16, kk * 32);
      v16h bf = load_frag_nk(sHi, 64, nt * 16, kk * 32);
      acc = wmma32(a, bf, acc);
    }
    const _Float16* w2p = w2 + (hy * 2 + (nt >> 1)) * 4096;
    for (int kk = 0; kk < 2; ++kk) {     // lo (upsample) chain, K = 64
      v16h a = load_frag_mk(w2p, 64, mt * 16, kk * 32);
      v16h bf = load_frag_nk(sLo, 64, (nt & 1) * 16, kk * 32);
      acc = wmma32(a, bf, acc);
    }
    const int col = lane & 15, kh = lane >> 4;
#pragma unroll
    for (int r = 0; r < 8; ++r) {
      int o = mt * 16 + r + 8 * kh;
      int s = nt * 16 + col;
      int par = s >> 5;
      int w = w0 + 2 * (s & 31) + par;
      out[(((long)b * 64 + o) * 384 + h) * 384 + w] = acc[r] + bias[o];
    }
  }
}

// ---------------------------------------------------------------------------
extern "C" void kernel_launch(void* const* d_in, const int* in_sizes, int n_in,
                              void* d_out, int out_size, void* d_ws, size_t ws_size,
                              hipStream_t stream) {
  const float* x      = (const float*)d_in[0];
  const float* wq_hi  = (const float*)d_in[1];
  const float* wk_hi  = (const float*)d_in[2];
  const float* wv_hi  = (const float*)d_in[3];
  const float* wq_lo  = (const float*)d_in[4];
  const float* wk_lo  = (const float*)d_in[5];
  const float* wv_lo  = (const float*)d_in[6];
  const float* temp   = (const float*)d_in[7];
  const float* ct_w   = (const float*)d_in[8];
  const float* ct_b   = (const float*)d_in[9];
  const float* po_w   = (const float*)d_in[10];
  float* out = (float*)d_out;

  // workspace layout
  const long N_LO  = 2L * 64 * 192 * 192;   //  4,718,592 floats
  const long N_HI  = 2L * 64 * 384 * 384;   // 18,874,368 floats
  float*    x_lo   = (float*)d_ws;
  float*    img_hi = x_lo + N_LO;
  float*    img_lo = img_hi + N_HI;
  _Float16* w16    = (_Float16*)(img_lo + N_LO);   // [6][256*256] f16
  _Float16* w1     = w16 + 6L * 65536;             // [64][64]
  _Float16* w2     = w1 + 4096;                    // [4][64][64]
  float*    bias   = (float*)(w2 + 4 * 4096);      // [64]

  // 1) weight conversion + epilogue folding
  {
    int total = 6 * 65536 + 4096 + 16384 + 64;
    prep_kernel<<<(total + 255) / 256, 256, 0, stream>>>(
        wq_hi, wk_hi, wv_hi, wq_lo, wk_lo, wv_lo, ct_w, ct_b, po_w,
        w16, w1, w2, bias);
  }
  // 2) maxpool
  maxpool_kernel<<<(int)(N_LO / 256), 256, 0, stream>>>(x, x_lo);
  // 3) hi branch attention: 2 * 24*24 tokens
  attn_kernel<<<2 * 24 * 24, 512, 0, stream>>>(
      x, w16 + 0L * 65536, w16 + 1L * 65536, w16 + 2L * 65536,
      temp, img_hi, 384, 384, 24, 24);
  // 4) lo branch attention: 2 * 12*12 tokens
  attn_kernel<<<2 * 12 * 12, 512, 0, stream>>>(
      x_lo, w16 + 3L * 65536, w16 + 4L * 65536, w16 + 5L * 65536,
      temp, img_lo, 192, 192, 12, 12);
  // 5) fused ConvTranspose + concat + 1x1 conv
  epilogue_kernel<<<2 * 384 * 6, 256, 0, stream>>>(img_hi, img_lo, w1, w2, bias, out);
}
